// MambaBlock_67972152427238
// MI455X (gfx1250) — matrix operand order repaired and checked by reading, not compiled, not run on hardware
//
#include <hip/hip_runtime.h>
#include <cstdint>
#include <cstddef>

// ---------------------------------------------------------------------------
// Mamba block for MI455X (gfx1250, wave32, WMMA).
// cvt->f16, GEMM_in (WMMA 2x2, fused SiLU on z-half), conv+SiLU,
// GEMM_dbc (WMMA 2x2, contiguous branch-free epilogue), dt-proj+softplus,
// selective scan (VALU/v_exp, fused gate), GEMM_out (WMMA 2x2).
// ---------------------------------------------------------------------------

typedef _Float16 v16h __attribute__((ext_vector_type(16)));
typedef _Float16 v8h  __attribute__((ext_vector_type(8)));
typedef _Float16 v4h  __attribute__((ext_vector_type(4)));
typedef float    v8f  __attribute__((ext_vector_type(8)));

#define M_TOTAL   32768   // B*L = 8*4096
#define L_SEQ     4096
#define DMODEL    128
#define DINNER    256
#define DSTATE    128
#define DTRANK    8
#define N_XZ      512
#define N_DBC     264
#define N_DBC_PAD 288     // pad to multiple of 32 (wave n-tile) ; row stride of dbc
#define OFF_B     8       // dbc row offset of B (floats)  -> 32 B aligned
#define OFF_C     136     // dbc row offset of C (floats)  -> 544 B aligned

// ---------------- fast transcendental helpers (v_rcp/v_exp/v_log) ----------
__device__ __forceinline__ float fast_sigmoid(float x) {
  return __builtin_amdgcn_rcpf(1.0f + __expf(-x));   // v_rcp_f32, no div macro
}
__device__ __forceinline__ float silu_f(float v) { return v * fast_sigmoid(v); }

// ---------------- f32 -> f16 conversion (x4 vectorized) --------------------
__global__ void k_cvt_f16x4(const float* __restrict__ src, _Float16* __restrict__ dst, int n4) {
  int i = blockIdx.x * blockDim.x + threadIdx.x;
  if (i < n4) {
    float4 v = ((const float4*)src)[i];
    v4h h = { (_Float16)v.x, (_Float16)v.y, (_Float16)v.z, (_Float16)v.w };
    ((v4h*)dst)[i] = h;
  }
}

// W_x is [264,256]; pad N to 288 rows (zeros) so 32-col wave tiles fit.
__global__ void k_cvt_wx(const float* __restrict__ src, _Float16* __restrict__ dst) {
  int i = blockIdx.x * blockDim.x + threadIdx.x;   // over 288*256 exactly
  int n = i >> 8, k = i & 255;
  float v = (n < N_DBC) ? src[n * 256 + k] : 0.0f;
  dst[i] = (_Float16)v;
}

// ---------------- 2x2 register-blocked WMMA MAC ----------------------------
// A: [M][K] f16 row-major. Bw: [N][K] f16 row-major (weight rows over K).
// Wave computes 32x32 output = 4 x (16x16) tiles; A/B frags reused 2x each.
__device__ __forceinline__ void wmma_2x2(const _Float16* __restrict__ A,
                                         const _Float16* __restrict__ Bw,
                                         int m0, int n0, int K, int lane,
                                         v8f c[2][2]) {
  const int half = lane >> 4;
  const int lid  = lane & 15;
  const _Float16* ar0 = A  + (size_t)(m0 + lid) * K;
  const _Float16* ar1 = ar0 + (size_t)16 * K;
  const _Float16* br0 = Bw + (size_t)(n0 + lid) * K;
  const _Float16* br1 = br0 + (size_t)16 * K;
#pragma unroll
  for (int k0 = 0; k0 < K; k0 += 32) {
    // A frag: K = 8*half + e (e<8), 16 + 8*half + (e-8) (e>=8): 2x16B chunks
    v8h a0l = *(const v8h*)(ar0 + k0 + 8 * half);
    v8h a0h = *(const v8h*)(ar0 + k0 + 16 + 8 * half);
    v8h a1l = *(const v8h*)(ar1 + k0 + 8 * half);
    v8h a1h = *(const v8h*)(ar1 + k0 + 16 + 8 * half);
    // B frag: same K order as A: 2x16B chunks
    v8h b0l = *(const v8h*)(br0 + k0 + 8 * half);
    v8h b0h = *(const v8h*)(br0 + k0 + 16 + 8 * half);
    v8h b1l = *(const v8h*)(br1 + k0 + 8 * half);
    v8h b1h = *(const v8h*)(br1 + k0 + 16 + 8 * half);
    v16h a0 = __builtin_shufflevector(a0l, a0h, 0,1,2,3,4,5,6,7,8,9,10,11,12,13,14,15);
    v16h a1 = __builtin_shufflevector(a1l, a1h, 0,1,2,3,4,5,6,7,8,9,10,11,12,13,14,15);
    v16h b0 = __builtin_shufflevector(b0l, b0h, 0,1,2,3,4,5,6,7,8,9,10,11,12,13,14,15);
    v16h b1 = __builtin_shufflevector(b1l, b1h, 0,1,2,3,4,5,6,7,8,9,10,11,12,13,14,15);
    c[0][0] = __builtin_amdgcn_wmma_f32_16x16x32_f16(false, a0, false, b0, (short)0, c[0][0], false, false);
    c[0][1] = __builtin_amdgcn_wmma_f32_16x16x32_f16(false, a0, false, b1, (short)0, c[0][1], false, false);
    c[1][0] = __builtin_amdgcn_wmma_f32_16x16x32_f16(false, a1, false, b0, (short)0, c[1][0], false, false);
    c[1][1] = __builtin_amdgcn_wmma_f32_16x16x32_f16(false, a1, false, b1, (short)0, c[1][1], false, false);
  }
}

__device__ __forceinline__ void zero_c(v8f c[2][2]) {
#pragma unroll
  for (int i = 0; i < 2; ++i)
#pragma unroll
    for (int j = 0; j < 2; ++j) c[i][j] = (v8f){0.f,0.f,0.f,0.f,0.f,0.f,0.f,0.f};
}

// ---------------- GEMM 1: xz = u @ W_in^T  (K=128, N=512) ------------------
// Stores one [m][512] buffer; SiLU applied on cols >= 256 (tile-uniform).
__global__ void k_gemm_in(const _Float16* __restrict__ u16,
                          const _Float16* __restrict__ w16,
                          float* __restrict__ xz) {
  int tid = threadIdx.x, wave = tid >> 5, lane = tid & 31;
  int m0 = blockIdx.y * 128 + (wave & 3) * 32;
  int n0 = blockIdx.x * 64  + (wave >> 2) * 32;
  v8f c[2][2]; zero_c(c);
  wmma_2x2(u16, w16, m0, n0, DMODEL, lane, c);
  int lid = lane & 15, half = lane >> 4;
#pragma unroll
  for (int mi = 0; mi < 2; ++mi)
#pragma unroll
    for (int ni = 0; ni < 2; ++ni) {
      int tb    = n0 + 16 * ni;                 // tile base col: wave-uniform
      int col   = tb + lid;
      int mbase = m0 + 16 * mi + 8 * half;
      if (tb < DINNER) {                        // x-half: raw store
#pragma unroll
        for (int r = 0; r < 8; ++r)
          xz[(size_t)(mbase + r) * N_XZ + col] = c[mi][ni][r];
      } else {                                  // z-half: fused SiLU
#pragma unroll
        for (int r = 0; r < 8; ++r)
          xz[(size_t)(mbase + r) * N_XZ + col] = silu_f(c[mi][ni][r]);
      }
    }
}

// ---------------- causal depthwise conv(4) + SiLU --------------------------
__global__ void k_conv(const float* __restrict__ xz, const float* __restrict__ cw,
                       const float* __restrict__ cb, float* __restrict__ xc,
                       _Float16* __restrict__ xc16) {
  int m = blockIdx.x;          // row in [0, B*L)
  int d = threadIdx.x;         // channel in [0, 256)
  int l = m & (L_SEQ - 1);     // position within batch
  float acc = cb[d];
#pragma unroll
  for (int j = 0; j < 4; ++j)  // xc[l] = sum_k w[k]*xi[l+k-3], j = 3-k
    if (l >= j) acc += cw[d * 4 + (3 - j)] * xz[(size_t)(m - j) * N_XZ + d];
  float s = silu_f(acc);
  size_t idx = (size_t)m * DINNER + d;
  xc[idx]   = s;
  xc16[idx] = (_Float16)s;
}

// ---------------- GEMM 2: dbc = xc @ W_x^T  (K=256, N pad 288) -------------
// Branch-free epilogue: contiguous [m][288] store (dt@0, B@8, C@136).
__global__ void k_gemm_dbc(const _Float16* __restrict__ a16,
                           const _Float16* __restrict__ w16,
                           float* __restrict__ dbc) {
  int tid = threadIdx.x, wave = tid >> 5, lane = tid & 31;
  int m0 = blockIdx.y * 128 + (wave & 3) * 32;
  int n0 = blockIdx.x * 64  + (wave >> 2) * 32;
  if (n0 >= N_DBC_PAD) return;          // wave-uniform skip of overhang tiles
  v8f c[2][2]; zero_c(c);
  wmma_2x2(a16, w16, m0, n0, DINNER, lane, c);
  int lid = lane & 15, half = lane >> 4;
#pragma unroll
  for (int mi = 0; mi < 2; ++mi)
#pragma unroll
    for (int ni = 0; ni < 2; ++ni) {
      int col   = n0 + 16 * ni + lid;
      int mbase = m0 + 16 * mi + 8 * half;
#pragma unroll
      for (int r = 0; r < 8; ++r)
        dbc[(size_t)(mbase + r) * N_DBC_PAD + col] = c[mi][ni][r];
    }
}

// ---- delta = softplus(dt @ W_dt^T + b_dt); dd = {delta, delta*xc} ---------
__global__ void k_delta(const float* __restrict__ dbc, const float* __restrict__ wdt,
                        const float* __restrict__ bdt, const float* __restrict__ xc,
                        float2* __restrict__ dd) {
  int m = blockIdx.x, d = threadIdx.x;
  float acc = bdt[d];
#pragma unroll
  for (int r = 0; r < DTRANK; ++r)
    acc += dbc[(size_t)m * N_DBC_PAD + r] * wdt[d * DTRANK + r];
  float sp = (acc > 20.0f) ? acc : __logf(1.0f + __expf(acc));  // v_log/v_exp
  float xcv = xc[(size_t)m * DINNER + d];
  dd[(size_t)m * DINNER + d] = make_float2(sp, sp * xcv);
}

// ---------------- selective scan (sequential over L) -----------------------
// One wave per (b, d): 128 states = 4 per lane in VGPRs. 2048 waves total.
// Writes y16 = f16((y_scan + xc*D) * silu(z)) directly (epilogue fused).
__global__ void k_scan(const float2* __restrict__ dd, const float* __restrict__ dbc,
                       const float* __restrict__ xc, const float* __restrict__ xz,
                       const float* __restrict__ A_log, const float* __restrict__ Dp,
                       _Float16* __restrict__ y16) {
  int tid  = threadIdx.x, wave = tid >> 5, lane = tid & 31;
  int b    = blockIdx.x >> 5;                 // 8 batches
  int d    = (blockIdx.x & 31) * 8 + wave;    // 256 channels
  int s0   = 4 * lane;                        // this lane's first state
  // A = -exp(A_log): fold once per lane (4 states)
  float a0 = -__expf(A_log[d * DSTATE + s0 + 0]);
  float a1 = -__expf(A_log[d * DSTATE + s0 + 1]);
  float a2 = -__expf(A_log[d * DSTATE + s0 + 2]);
  float a3 = -__expf(A_log[d * DSTATE + s0 + 3]);
  float Dd = Dp[d];
  float h0 = 0.f, h1 = 0.f, h2 = 0.f, h3 = 0.f;
  size_t mrow = (size_t)b * L_SEQ;
  for (int l = 0; l < L_SEQ; ++l) {
    size_t m = mrow + (size_t)l;
    const float* row = dbc + m * N_DBC_PAD;
    float4 Bv = *(const float4*)(row + OFF_B + s0);   // one b128 load
    float4 Cv = *(const float4*)(row + OFF_C + s0);   // one b128 load
    float2 dv = dd[m * DINNER + d];                   // one b64 broadcast
    float dl = dv.x, du = dv.y;
    h0 = h0 * __expf(a0 * dl) + du * Bv.x;
    h1 = h1 * __expf(a1 * dl) + du * Bv.y;
    h2 = h2 * __expf(a2 * dl) + du * Bv.z;
    h3 = h3 * __expf(a3 * dl) + du * Bv.w;
    float acc = h0 * Cv.x + h1 * Cv.y + h2 * Cv.z + h3 * Cv.w;
#pragma unroll
    for (int off = 16; off >= 1; off >>= 1)   // wave32 xor reduction
      acc += __shfl_xor(acc, off, 32);
    if (lane == 0) {
      float gate = xz[m * N_XZ + DINNER + d];         // silu(z), precomputed
      float yv   = (acc + xc[m * DINNER + d] * Dd) * gate;
      y16[m * DINNER + d] = (_Float16)yv;
    }
    if (l + 1 < L_SEQ) {                      // global_prefetch_b8 next step
      __builtin_prefetch(row + N_DBC_PAD + OFF_B + s0, 0, 0);
      __builtin_prefetch(row + N_DBC_PAD + OFF_C + s0, 0, 0);
    }
  }
}

// ---------------- GEMM 3: out = y @ W_out^T  (K=256, N=128) ----------------
__global__ void k_gemm_out(const _Float16* __restrict__ y16,
                           const _Float16* __restrict__ w16,
                           float* __restrict__ out) {
  int tid = threadIdx.x, wave = tid >> 5, lane = tid & 31;
  int m0 = blockIdx.y * 128 + (wave & 3) * 32;
  int n0 = blockIdx.x * 64  + (wave >> 2) * 32;
  v8f c[2][2]; zero_c(c);
  wmma_2x2(y16, w16, m0, n0, DINNER, lane, c);
  int lid = lane & 15, half = lane >> 4;
#pragma unroll
  for (int mi = 0; mi < 2; ++mi)
#pragma unroll
    for (int ni = 0; ni < 2; ++ni) {
      int col   = n0 + 16 * ni + lid;
      int mbase = m0 + 16 * mi + 8 * half;
#pragma unroll
      for (int r = 0; r < 8; ++r)
        out[(size_t)(mbase + r) * DMODEL + col] = c[mi][ni][r];
    }
}

// ---------------------------------------------------------------------------
extern "C" void kernel_launch(void* const* d_in, const int* in_sizes, int n_in,
                              void* d_out, int out_size, void* d_ws, size_t ws_size,
                              hipStream_t stream) {
  const float* x      = (const float*)d_in[0];   // [8,128,64,64] == [32768,128]
  const float* W_in   = (const float*)d_in[1];   // [512,128]
  const float* conv_w = (const float*)d_in[2];   // [256,4]
  const float* conv_b = (const float*)d_in[3];   // [256]
  const float* W_x    = (const float*)d_in[4];   // [264,256]
  const float* W_dt   = (const float*)d_in[5];   // [256,8]
  const float* b_dt   = (const float*)d_in[6];   // [256]
  const float* A_log  = (const float*)d_in[7];   // [256,128]
  const float* Dp     = (const float*)d_in[8];   // [256]
  const float* W_out  = (const float*)d_in[9];   // [128,256]
  float* out = (float*)d_out;

  // ---- workspace layout (bump alloc, ~231 MB, offsets 256B-aligned) ----
  char* ws = (char*)d_ws;
  size_t off = 0;
  auto alloc = [&](size_t bytes) { char* p = ws + off; off += (bytes + 255) & ~(size_t)255; return p; };
  _Float16* Win16  = (_Float16*)alloc((size_t)512 * 128 * 2);
  _Float16* Wx16   = (_Float16*)alloc((size_t)N_DBC_PAD * 256 * 2);
  _Float16* Wout16 = (_Float16*)alloc((size_t)128 * 256 * 2);
  _Float16* u16    = (_Float16*)alloc((size_t)M_TOTAL * 128 * 2);
  float*    xz     = (float*)   alloc((size_t)M_TOTAL * N_XZ * 4);
  float*    xc     = (float*)   alloc((size_t)M_TOTAL * 256 * 4);
  _Float16* xc16   = (_Float16*)alloc((size_t)M_TOTAL * 256 * 2);
  float*    dbc    = (float*)   alloc((size_t)M_TOTAL * N_DBC_PAD * 4);
  float2*   dd     = (float2*)  alloc((size_t)M_TOTAL * 256 * 8);
  _Float16* y16 = xc16;        // xc16 dead after k_gemm_dbc

  // weight + activation f16 conversion (sizes all divisible by 4)
  k_cvt_f16x4<<<(512 * 128 / 4 + 255) / 256, 256, 0, stream>>>(W_in, Win16, 512 * 128 / 4);
  k_cvt_f16x4<<<(128 * 256 / 4 + 255) / 256, 256, 0, stream>>>(W_out, Wout16, 128 * 256 / 4);
  k_cvt_f16x4<<<(M_TOTAL * 128 / 4 + 255) / 256, 256, 0, stream>>>(x, u16, M_TOTAL * 128 / 4);
  k_cvt_wx <<<N_DBC_PAD, 256, 0, stream>>>(W_x, Wx16);

  // xz = u @ W_in^T (SiLU fused on z-half)   block 128x64, wave 32x32
  k_gemm_in <<<dim3(N_XZ / 64, M_TOTAL / 128), 256, 0, stream>>>(u16, Win16, xz);
  // causal conv + SiLU
  k_conv    <<<M_TOTAL, 256, 0, stream>>>(xz, conv_w, conv_b, xc, xc16);
  // dbc = xc @ W_x^T, contiguous [m][288]    (5 blocks cover 320, guard 288)
  k_gemm_dbc<<<dim3(5, M_TOTAL / 128), 256, 0, stream>>>(xc16, Wx16, dbc);
  // delta + delta*xc (interleaved float2)
  k_delta   <<<M_TOTAL, 256, 0, stream>>>(dbc, W_dt, b_dt, xc, dd);
  // selective scan (fused gate epilogue) -> y16
  k_scan    <<<256, 256, 0, stream>>>(dd, dbc, xc, xz, A_log, Dp, y16);
  // out = y @ W_out^T
  k_gemm_out<<<dim3(DMODEL / 64, M_TOTAL / 128), 256, 0, stream>>>(y16, Wout16, out);

  (void)in_sizes; (void)n_in; (void)out_size; (void)ws_size;
}
